// Emb_Split_Model_11166914970053
// MI455X (gfx1250) — matrix operand order repaired and checked by reading, not compiled, not run on hardware
//
#include <hip/hip_runtime.h>
#include <math.h>

typedef __attribute__((ext_vector_type(16))) _Float16 v16h;
typedef __attribute__((ext_vector_type(8)))  float    v8f;

#define H 256

// pack two f32 into one dword of f16 halves (v_cvt_pk_rtz_f16_f32)
__device__ __forceinline__ unsigned pack_f16(float a, float b)
{
  auto p = __builtin_amdgcn_cvt_pkrtz(a, b);   // __fp16 ext_vector(2)
  return __builtin_bit_cast(unsigned, p);
}

// ============================================================================
// WMMA GEMM: C[M,N] = A[M,K] @ B[K,N] (+bias)(+relu)
// Requires N % 64 == 0, K % 32 == 0 (true for all call sites). M arbitrary
// (out-of-range tile rows are clamped on load, guarded on store).
// 256 threads = 8 waves; block computes a 64x64 tile; each wave owns a 16x32
// output strip (two v_wmma_f32_16x16x32_f16 accumulators).
//
// Tiles are staged into LDS *in WMMA fragment order*: each lane's 16 f16
// operand halves are contiguous (32B), so fragment loads are 2x ds_load_b128.
//   A 16x32 sub-tile: lane = (r&15)|(hi<<4), hi=(k>>3)&1, i=(k&7)+((k&16)>>1)
//   B 32x16 sub-tile: lane = (n&15)|(hi<<4), hi=(k>>3)&1, i=(k&7)+((k&16)>>1)
// Staging is split into a batched-load phase (clause-able, multiple loads in
// flight) and a pack/store phase using v_cvt_pk_rtz_f16_f32.
// ============================================================================
__global__ __launch_bounds__(256) void k_wmma_gemm(
    const float* __restrict__ A, const float* __restrict__ Bm,
    float* __restrict__ C, const float* __restrict__ bias,
    int M, int N, int K, int relu)
{
  // 4 sub-tiles x 32 lanes x 8 dwords (16 f16 halves) = 4KB each
  __shared__ __align__(32) unsigned AsU[4 * 32 * 8];
  __shared__ __align__(32) unsigned BsU[4 * 32 * 8];

  const int tid  = threadIdx.x;
  const int wave = tid >> 5;
  const int lane = tid & 31;
  const int hl   = lane & 15;
  const int hi   = lane >> 4;
  const int m0 = blockIdx.y * 64;
  const int n0 = blockIdx.x * 64;
  const int wm = (wave & 3) * 16;  // wave's M offset in tile
  const int wn = (wave >> 2) * 32; // wave's N offset in tile (2 sub-tiles)

  // Precompute this thread's staging slots (loop-invariant).
  // A: idx j covers element (r, c..c+1) of the 64x32 tile.
  int aAddr[4], aLds[4], aCol[4];
  #pragma unroll
  for (int j = 0; j < 4; ++j) {
    int idx = tid + j * 256;
    int r = idx >> 4;              // 0..63
    int c = (idx & 15) * 2;        // 0,2,..,30
    int gr = m0 + r; if (gr > M - 1) gr = M - 1;   // clamp (stores are guarded)
    aAddr[j] = gr;
    aCol[j]  = c;
    int h2  = (c >> 3) & 1;
    int i2  = (c & 7) + ((c & 16) >> 1);
    int ln  = (r & 15) + (h2 << 4);
    aLds[j] = (r >> 4) * 256 + ln * 8 + (i2 >> 1);
  }

  // B: idx j covers elements (k,c),(k+1,c) of the 32x64 tile.
  int bK[4], bC[4], bLds[4];
  #pragma unroll
  for (int j = 0; j < 4; ++j) {
    int idx = tid + j * 256;
    int k = (idx >> 6) * 2;        // 0,2,..,30
    int c = idx & 63;
    bK[j] = k; bC[j] = c;
    int h2 = (k >> 3) & 1;
    int i2 = (k & 7) + ((k & 16) >> 1);
    int ln = (c & 15) + (h2 << 4);
    bLds[j] = (c >> 4) * 256 + ln * 8 + (i2 >> 1);
  }

  v8f acc0 = {}; v8f acc1 = {};

  for (int k0 = 0; k0 < K; k0 += 32) {
    // ---- load phase: all global loads issued back-to-back ----
    float2 va[4];
    #pragma unroll
    for (int j = 0; j < 4; ++j)
      va[j] = *(const float2*)(A + (size_t)aAddr[j] * K + k0 + aCol[j]);
    float vb0[4], vb1[4];
    #pragma unroll
    for (int j = 0; j < 4; ++j) {
      vb0[j] = Bm[(size_t)(k0 + bK[j])     * N + n0 + bC[j]];
      vb1[j] = Bm[(size_t)(k0 + bK[j] + 1) * N + n0 + bC[j]];
    }
    // ---- pack + LDS store phase (fragment order) ----
    #pragma unroll
    for (int j = 0; j < 4; ++j)
      AsU[aLds[j]] = pack_f16(va[j].x, va[j].y);
    #pragma unroll
    for (int j = 0; j < 4; ++j)
      BsU[bLds[j]] = pack_f16(vb0[j], vb1[j]);
    __syncthreads();

    // fragment loads: contiguous 32B per lane -> 2x ds_load_b128 each
    v16h af  = ((const v16h*)AsU)[(wave & 3) * 32 + lane];
    int  sb  = (wave >> 2) * 2;
    v16h bf0 = ((const v16h*)BsU)[sb * 32 + lane];
    v16h bf1 = ((const v16h*)BsU)[sb * 32 + 32 + lane];

    acc0 = __builtin_amdgcn_wmma_f32_16x16x32_f16(false, af, false, bf0,
                                                  (short)0, acc0, false, false);
    acc1 = __builtin_amdgcn_wmma_f32_16x16x32_f16(false, af, false, bf1,
                                                  (short)0, acc1, false, false);
    __syncthreads();
  }

  // C/D layout: VGPR r -> M = r (lanes 0-15) / M = 8+r (lanes 16-31), N = hl
  #pragma unroll
  for (int r = 0; r < 8; ++r) {
    int row = m0 + wm + (hi ? 8 : 0) + r;
    if (row < M) {
      int c0 = n0 + wn + hl, c1 = c0 + 16;
      float v0 = acc0[r], v1 = acc1[r];
      if (bias) { v0 += bias[c0]; v1 += bias[c1]; }
      if (relu) { v0 = fmaxf(v0, 0.f); v1 = fmaxf(v1, 0.f); }
      C[(size_t)row * N + c0] = v0;
      C[(size_t)row * N + c1] = v1;
    }
  }
}

// ============================================================================
// wave-per-row helpers (wave32)
// ============================================================================
__global__ void k_rowdot(const float* __restrict__ h, const float* __restrict__ a,
                         float* __restrict__ o, int rows)
{
  int row  = blockIdx.x * (blockDim.x >> 5) + (threadIdx.x >> 5);
  int lane = threadIdx.x & 31;
  if (row >= rows) return;
  const float* hp = h + (size_t)row * H;
  float s = 0.f;
  for (int i = lane; i < H; i += 32) s += hp[i] * a[i];
  for (int off = 16; off; off >>= 1) s += __shfl_xor(s, off, 32);
  if (lane == 0) o[row] = s;
}

__global__ void k_l2norm(const float* __restrict__ x, float* __restrict__ y, int rows)
{
  int row  = blockIdx.x * (blockDim.x >> 5) + (threadIdx.x >> 5);
  int lane = threadIdx.x & 31;
  if (row >= rows) return;
  const float* xp = x + (size_t)row * H;
  float s = 0.f;
  for (int i = lane; i < H; i += 32) { float v = xp[i]; s += v * v; }
  for (int off = 16; off; off >>= 1) s += __shfl_xor(s, off, 32);
  float inv = 1.f / fmaxf(sqrtf(s), 1e-12f);
  float* yp = y + (size_t)row * H;
  for (int i = lane; i < H; i += 32) yp[i] = xp[i] * inv;
}

// ============================================================================
// fills
// ============================================================================
__global__ void k_fill_f32(float* p, float v, long long n)
{
  long long i = blockIdx.x * (long long)blockDim.x + threadIdx.x;
  if (i < n) p[i] = v;
}
__global__ void k_fill_u32(unsigned* p, unsigned v, long long n)
{
  long long i = blockIdx.x * (long long)blockDim.x + threadIdx.x;
  if (i < n) p[i] = v;
}

// ============================================================================
// segment softmax over edges (3 passes, atomics). Float-max via ordered uints.
// selfloop=1 means src=dst=edge index (p-p self loops appended after edge_pp).
// ============================================================================
__device__ __forceinline__ unsigned ford(float f)
{
  unsigned u = __float_as_uint(f);
  return (u & 0x80000000u) ? ~u : (u | 0x80000000u);
}
__device__ __forceinline__ float funord(unsigned u)
{
  return (u & 0x80000000u) ? __uint_as_float(u & 0x7FFFFFFFu) : __uint_as_float(~u);
}

__global__ void k_edge_max(const int* __restrict__ src, const int* __restrict__ dst,
                           int E, int selfloop,
                           const float* __restrict__ als, const float* __restrict__ ald,
                           float* __restrict__ ebuf, unsigned* __restrict__ mord)
{
  int e = blockIdx.x * blockDim.x + threadIdx.x;
  if (e >= E) return;
  int s = selfloop ? e : src[e];
  int d = selfloop ? e : dst[e];
  float x = als[s] + ald[d];
  x = (x > 0.f) ? x : 0.2f * x;           // leaky_relu(0.2)
  ebuf[e] = x;
  atomicMax(mord + d, ford(x));
}

__global__ void k_edge_exp(const int* __restrict__ dst, int E, int selfloop,
                           const float* __restrict__ ebuf, const unsigned* __restrict__ mord,
                           float* __restrict__ exbuf, float* __restrict__ denom)
{
  int e = blockIdx.x * blockDim.x + threadIdx.x;
  if (e >= E) return;
  int d = selfloop ? e : dst[e];
  float m = funord(mord[d]);
  if (!isfinite(m)) m = 0.f;
  float ex = expf(ebuf[e] - m);
  exbuf[e] = ex;
  atomicAdd(denom + d, ex);
}

__global__ void k_edge_scatter(const int* __restrict__ src, const int* __restrict__ dst,
                               int E, int selfloop,
                               const float* __restrict__ exbuf, const float* __restrict__ denom,
                               const float* __restrict__ hsrc, float* __restrict__ out)
{
  long long t = blockIdx.x * (long long)blockDim.x + threadIdx.x;
  int e  = (int)(t >> 6);                 // 64 threads per edge, 4 channels each
  int c4 = (int)(t & 63) * 4;
  if (e >= E) return;
  int s = selfloop ? e : src[e];
  int d = selfloop ? e : dst[e];
  float coef = exbuf[e] / (denom[d] + 1e-16f);
  float4 hv = *(const float4*)(hsrc + (size_t)s * H + c4);
  float* op = out + (size_t)d * H + c4;
  atomicAdd(op + 0, coef * hv.x);
  atomicAdd(op + 1, coef * hv.y);
  atomicAdd(op + 2, coef * hv.z);
  atomicAdd(op + 3, coef * hv.w);
}

__global__ void k_bias_act(float* __restrict__ x, const float* __restrict__ b,
                           long long total, int relu)
{
  long long i = blockIdx.x * (long long)blockDim.x + threadIdx.x;
  if (i >= total) return;
  float v = x[i] + b[i & (H - 1)];
  x[i] = relu ? fmaxf(v, 0.f) : v;
}

// ============================================================================
// MLP head helpers
// ============================================================================
__global__ void k_gather_hid(const float* __restrict__ xdn, const float* __restrict__ xcn,
                             const int* __restrict__ d1, const int* __restrict__ d2,
                             const int* __restrict__ cc, float* __restrict__ hid, int Bq)
{
  long long t = blockIdx.x * (long long)blockDim.x + threadIdx.x;
  long long total = (long long)Bq * 768;
  if (t >= total) return;
  int r = (int)(t / 768), c = (int)(t % 768);
  float v;
  if (c < 256)      v = xdn[(size_t)d1[r] * H + c];
  else if (c < 512) v = xdn[(size_t)d2[r] * H + (c - 256)];
  else              v = xcn[(size_t)cc[r] * H + (c - 512)];
  hid[t] = v;
}

__global__ void k_gemm_smallN(const float* __restrict__ A, const float* __restrict__ W,
                              const float* __restrict__ b, float* __restrict__ out,
                              int M, int K, int N)
{
  int idx = blockIdx.x * blockDim.x + threadIdx.x;
  if (idx >= M * N) return;
  int r = idx / N, n = idx % N;
  float s = b[n];
  const float* ap = A + (size_t)r * K;
  for (int k = 0; k < K; ++k) s += ap[k] * W[(size_t)k * N + n];
  out[idx] = s;
}

// ============================================================================
// host orchestration
// ============================================================================
static void gemm(const float* A, const float* Bm, float* C, const float* bias,
                 int M, int N, int K, int relu, hipStream_t s)
{
  dim3 g(N / 64, (M + 63) / 64), b(256);
  k_wmma_gemm<<<g, b, 0, s>>>(A, Bm, C, bias, M, N, K, relu);
}

extern "C" void kernel_launch(void* const* d_in, const int* in_sizes, int n_in,
                              void* d_out, int out_size, void* d_ws, size_t ws_size,
                              hipStream_t stream)
{
  (void)in_sizes; (void)n_in; (void)out_size; (void)ws_size;

  const float* drug_emb    = (const float*)d_in[0];
  const float* protein_emb = (const float*)d_in[1];
  const float* cell_emb    = (const float*)d_in[2];
  const float* W_pp = (const float*)d_in[3];
  const float* as_pp = (const float*)d_in[4];
  const float* ad_pp = (const float*)d_in[5];
  const float* b_pp  = (const float*)d_in[6];
  const float* W_dp = (const float*)d_in[7];
  const float* as_dp = (const float*)d_in[8];
  const float* ad_dp = (const float*)d_in[9];
  const float* b_dp  = (const float*)d_in[10];
  const float* W_cp = (const float*)d_in[11];
  const float* as_cp = (const float*)d_in[12];
  const float* ad_cp = (const float*)d_in[13];
  const float* b_cp  = (const float*)d_in[14];
  const float* W1 = (const float*)d_in[15];
  const float* b1 = (const float*)d_in[16];
  const float* W2 = (const float*)d_in[17];
  const float* b2 = (const float*)d_in[18];
  const float* W3 = (const float*)d_in[19];
  const float* b3 = (const float*)d_in[20];
  const int* edge_pp = (const int*)d_in[21];   // [2, 400000] (row0=src, row1=dst)
  const int* edge_dp = (const int*)d_in[22];   // [2, 60000]
  const int* edge_cp = (const int*)d_in[23];   // [2, 300000]
  const int* drug1 = (const int*)d_in[24];
  const int* drug2 = (const int*)d_in[25];
  const int* cellI = (const int*)d_in[26];

  const int NP = 19000, ND = 2000, NC = 100, Bq = 8192;
  const int Epp = 400000, Edp = 60000, Ecp = 300000;
  const size_t NPH = (size_t)NP * H, NDH = (size_t)ND * H, NCH = (size_t)NC * H;

  float* ws = (float*)d_ws;
  // persistent buffers
  float* x_p  = ws;                       // NPH
  float* x_d  = x_p + NPH;                // NDH
  float* x_c  = x_d + NDH;                // NCH
  float* xd_n = x_c + NCH;                // NDH
  float* xc_n = xd_n + NDH;               // NCH
  float* S    = xc_n + NCH;               // union scratch
  // GNN-phase layout inside S
  float*    h_a   = S;                    // NPH
  float*    h_b   = h_a + NPH;            // NDH
  float*    outb  = h_b + NDH;            // NPH (out_p / out_d / out_c)
  float*    als   = outb + NPH;           // NP
  float*    ald   = als + NP;             // NP
  unsigned* mord  = (unsigned*)(ald + NP);// NP
  float*    denom = (float*)(mord + NP);  // NP
  float*    ebuf  = denom + NP;           // Epp + NP
  float*    exbuf = ebuf + (Epp + NP);    // Epp + NP
  // MLP-phase layout inside S (after GNN scratch is dead)
  float* hid = S;                         // Bq*768
  float* h1  = hid + (size_t)Bq * 768;    // Bq*1536
  float* h2  = h1 + (size_t)Bq * 1536;    // Bq*512

  const unsigned NEG_INF_ORD = 0x007FFFFFu;  // ford(-inf)

  (void)hipMemcpyAsync(x_p, protein_emb, NPH * sizeof(float), hipMemcpyDeviceToDevice, stream);
  (void)hipMemcpyAsync(x_d, drug_emb,    NDH * sizeof(float), hipMemcpyDeviceToDevice, stream);
  (void)hipMemcpyAsync(x_c, cell_emb,    NCH * sizeof(float), hipMemcpyDeviceToDevice, stream);

  auto run_conv = [&](const float* xs, const float* xdst, int Ns, int Nd,
                      const float* Wl, const float* a_s, const float* a_d, const float* bl,
                      const int* esrc, const int* edst, int E, int add_self, int relu,
                      float* out) {
    gemm(xs, Wl, h_a, nullptr, Ns, H, H, 0, stream);
    const float* hsrc = h_a;
    const float* hdst = h_a;
    if (xdst != xs) {
      gemm(xdst, Wl, h_b, nullptr, Nd, H, H, 0, stream);
      hdst = h_b;
    }
    k_rowdot<<<(Ns + 7) / 8, 256, 0, stream>>>(hsrc, a_s, als, Ns);
    k_rowdot<<<(Nd + 7) / 8, 256, 0, stream>>>(hdst, a_d, ald, Nd);
    k_fill_u32<<<(Nd + 255) / 256, 256, 0, stream>>>(mord, NEG_INF_ORD, Nd);
    k_fill_f32<<<(Nd + 255) / 256, 256, 0, stream>>>(denom, 0.f, Nd);
    long long outN = (long long)Nd * H;
    k_fill_f32<<<(unsigned)((outN + 255) / 256), 256, 0, stream>>>(out, 0.f, outN);

    k_edge_max<<<(E + 255) / 256, 256, 0, stream>>>(esrc, edst, E, 0, als, ald, ebuf, mord);
    if (add_self)
      k_edge_max<<<(Ns + 255) / 256, 256, 0, stream>>>(nullptr, nullptr, Ns, 1, als, ald, ebuf + E, mord);

    k_edge_exp<<<(E + 255) / 256, 256, 0, stream>>>(edst, E, 0, ebuf, mord, exbuf, denom);
    if (add_self)
      k_edge_exp<<<(Ns + 255) / 256, 256, 0, stream>>>(nullptr, Ns, 1, ebuf + E, mord, exbuf + E, denom);

    long long T = (long long)E * 64;
    k_edge_scatter<<<(unsigned)((T + 255) / 256), 256, 0, stream>>>(esrc, edst, E, 0, exbuf, denom, hsrc, out);
    if (add_self) {
      T = (long long)Ns * 64;
      k_edge_scatter<<<(unsigned)((T + 255) / 256), 256, 0, stream>>>(nullptr, nullptr, Ns, 1, exbuf + E, denom, hsrc, out);
    }
    k_bias_act<<<(unsigned)((outN + 255) / 256), 256, 0, stream>>>(out, bl, outN, relu);
  };

  for (int l = 0; l < 2; ++l) {
    const size_t wo = (size_t)l * H * H, vo = (size_t)l * H;
    // p-p GAT (self loops, no relu) -> l2norm into x_p
    run_conv(x_p, x_p, NP, NP, W_pp + wo, as_pp + vo, ad_pp + vo, b_pp + vo,
             edge_pp, edge_pp + Epp, Epp, /*self*/1, /*relu*/0, outb);
    k_l2norm<<<(NP + 7) / 8, 256, 0, stream>>>(outb, x_p, NP);
    // d-p GAT (proteins -> drugs), relu
    run_conv(x_p, x_d, NP, ND, W_dp + wo, as_dp + vo, ad_dp + vo, b_dp + vo,
             edge_dp, edge_dp + Edp, Edp, 0, 1, outb);
    (void)hipMemcpyAsync(x_d, outb, NDH * sizeof(float), hipMemcpyDeviceToDevice, stream);
    // c-p GAT (proteins -> cells), relu
    run_conv(x_p, x_c, NP, NC, W_cp + wo, as_cp + vo, ad_cp + vo, b_cp + vo,
             edge_cp, edge_cp + Ecp, Ecp, 0, 1, outb);
    (void)hipMemcpyAsync(x_c, outb, NCH * sizeof(float), hipMemcpyDeviceToDevice, stream);
  }

  // MLP head
  k_l2norm<<<(ND + 7) / 8, 256, 0, stream>>>(x_d, xd_n, ND);
  k_l2norm<<<(NC + 7) / 8, 256, 0, stream>>>(x_c, xc_n, NC);
  long long hidN = (long long)Bq * 768;
  k_gather_hid<<<(unsigned)((hidN + 255) / 256), 256, 0, stream>>>(xd_n, xc_n, drug1, drug2, cellI, hid, Bq);
  gemm(hid, W1, h1, b1, Bq, 1536, 768, 1, stream);
  gemm(h1,  W2, h2, b2, Bq, 512, 1536, 1, stream);
  k_gemm_smallN<<<(Bq * 2 + 255) / 256, 256, 0, stream>>>(h2, W3, b3, (float*)d_out, Bq, 512, 2);
}
